// CausalSelfAttention_76424648065171
// MI455X (gfx1250) — matrix operand order repaired and checked by reading, not compiled, not run on hardware
//
#include <hip/hip_runtime.h>

// ---------------------------------------------------------------------------
// CDNA5 (gfx1250) causal self-attention, bf16 WMMA with fp32 accumulation.
// Wave32; WMMA 16x16x32 bf16 -> f32. GEMMs use rotation-based double buffering
// (two K-steps per iteration, fragments reloaded in place). sched_barriers
// keep only one fragment-set's loads in flight -> stays under 256 VGPRs.
// Attention is flash-style streaming softmax with in-place K/V reload.
// ---------------------------------------------------------------------------

typedef __attribute__((ext_vector_type(16))) __bf16 bf16x16;
typedef __attribute__((ext_vector_type(8)))  __bf16 bf16x8;
typedef __attribute__((ext_vector_type(8)))  float  f32x8;

#define Bb   4
#define Nn   2048
#define Dd   1024
#define Hh   16
#define HD   64
#define N3   3072   // 3*D

#if __has_builtin(__builtin_amdgcn_sched_barrier)
#define SCHED_FENCE() __builtin_amdgcn_sched_barrier(0)
#else
#define SCHED_FENCE()
#endif

// Build a 16-element bf16 fragment from two 16-byte chunks.
static __device__ __forceinline__ bf16x16 ld2x8(const __bf16* lo, const __bf16* hi) {
    bf16x8 a = *(const bf16x8*)lo;
    bf16x8 b = *(const bf16x8*)hi;
    return __builtin_shufflevector(a, b, 0,1,2,3,4,5,6,7,8,9,10,11,12,13,14,15);
}
// A-fragment: per-lane K = 16h..16h+15 contiguous within a 32-wide K slab
static __device__ __forceinline__ bf16x16 ld_a(const __bf16* row, int k0, int h2) {
    return ld2x8(row + k0 + 16 * h2, row + k0 + 16 * h2 + 8);
}
// B-fragment: per-lane column, K = 16h..16h+15 contiguous (ptr already includes 16h)
static __device__ __forceinline__ bf16x16 ld_b(const __bf16* p) {
    return ld2x8(p, p + 8);
}

// D = A(16x32 bf16) * B(32x16 bf16) + C(16x16 f32)
static __device__ __forceinline__ f32x8 wmma_bf16(bf16x16 a, bf16x16 b, f32x8 c) {
    return __builtin_amdgcn_wmma_f32_16x16x32_bf16(
        /*neg_a=*/false, a, /*neg_b=*/false, b,
        /*c_mod=*/(short)0, c, /*reuse_a=*/false, /*reuse_b=*/false);
}

// Half-wave (16-lane) reductions; rows of a C-frag live across 16 lanes.
static __device__ __forceinline__ float hmax16(float v) {
    v = fmaxf(v, __shfl_xor(v, 1, 16));
    v = fmaxf(v, __shfl_xor(v, 2, 16));
    v = fmaxf(v, __shfl_xor(v, 4, 16));
    v = fmaxf(v, __shfl_xor(v, 8, 16));
    return v;
}
static __device__ __forceinline__ float hsum16(float v) {
    v += __shfl_xor(v, 1, 16);
    v += __shfl_xor(v, 2, 16);
    v += __shfl_xor(v, 4, 16);
    v += __shfl_xor(v, 8, 16);
    return v;
}

// ---------------------------------------------------------------------------
// fp32 -> bf16 conversion (grid-stride)
// ---------------------------------------------------------------------------
__global__ void cvt_bf16_kernel(const float* __restrict__ s, __bf16* __restrict__ d, int n) {
    int i = blockIdx.x * blockDim.x + threadIdx.x;
    int stride = gridDim.x * blockDim.x;
    for (; i < n; i += stride) d[i] = (__bf16)s[i];
}

// ---------------------------------------------------------------------------
// Shared 32x64-tile GEMM core. Two K-steps per iteration; fragment sets
// reloaded in place. A sched fence after each half-iteration stops the
// scheduler from merging both load batches (register-pressure cap).
// ---------------------------------------------------------------------------
#define GEMM_CORE(arow0, arow1, bptr, acc0, acc1)                                  \
    {                                                                              \
        bf16x16 a0A = ld_a(arow0, 0, h2),  a1A = ld_a(arow1, 0, h2);               \
        bf16x16 a0B = ld_a(arow0, 32, h2), a1B = ld_a(arow1, 32, h2);              \
        bf16x16 bA[4], bB[4];                                                      \
        _Pragma("unroll")                                                          \
        for (int f = 0; f < 4; ++f) {                                              \
            bA[f] = ld_b(bptr + (size_t)f * 16 * Dd);                              \
            bB[f] = ld_b(bptr + (size_t)f * 16 * Dd + 32);                         \
        }                                                                          \
        int k = 64;                                                                \
        for (int it = 0; it < 15; ++it) {                                          \
            __builtin_prefetch(arow0 + k + 448, 0, 3);                             \
            _Pragma("unroll")                                                      \
            for (int f = 0; f < 4; ++f) acc0[f] = wmma_bf16(a0A, bA[f], acc0[f]);  \
            _Pragma("unroll")                                                      \
            for (int f = 0; f < 4; ++f) acc1[f] = wmma_bf16(a1A, bA[f], acc1[f]);  \
            a0A = ld_a(arow0, k, h2); a1A = ld_a(arow1, k, h2);                    \
            _Pragma("unroll")                                                      \
            for (int f = 0; f < 4; ++f) bA[f] = ld_b(bptr + (size_t)f*16*Dd + k);  \
            k += 32;                                                               \
            SCHED_FENCE();                                                         \
            _Pragma("unroll")                                                      \
            for (int f = 0; f < 4; ++f) acc0[f] = wmma_bf16(a0B, bB[f], acc0[f]);  \
            _Pragma("unroll")                                                      \
            for (int f = 0; f < 4; ++f) acc1[f] = wmma_bf16(a1B, bB[f], acc1[f]);  \
            a0B = ld_a(arow0, k, h2); a1B = ld_a(arow1, k, h2);                    \
            _Pragma("unroll")                                                      \
            for (int f = 0; f < 4; ++f) bB[f] = ld_b(bptr + (size_t)f*16*Dd + k);  \
            k += 32;                                                               \
            SCHED_FENCE();                                                         \
        }                                                                          \
        _Pragma("unroll")                                                          \
        for (int f = 0; f < 4; ++f) acc0[f] = wmma_bf16(a0A, bA[f], acc0[f]);      \
        _Pragma("unroll")                                                          \
        for (int f = 0; f < 4; ++f) acc1[f] = wmma_bf16(a1A, bA[f], acc1[f]);      \
        _Pragma("unroll")                                                          \
        for (int f = 0; f < 4; ++f) acc0[f] = wmma_bf16(a0B, bB[f], acc0[f]);      \
        _Pragma("unroll")                                                          \
        for (int f = 0; f < 4; ++f) acc1[f] = wmma_bf16(a1B, bB[f], acc1[f]);      \
    }

// ---------------------------------------------------------------------------
// QKV projection: qkv[m][n] = sum_k x[m][k]*w_qkv[n][k] + b_qkv[n]
// M=8192, N=3072, K=1024.  One wave computes a 32x64 tile.
// Q,K stored [B*H][N][HD] bf16; V stored transposed [B*H][HD][N] bf16.
// ---------------------------------------------------------------------------
__global__ __launch_bounds__(256) void qkv_kernel(
    const __bf16* __restrict__ xb, const __bf16* __restrict__ wb,
    const float* __restrict__ bias,
    __bf16* __restrict__ qb, __bf16* __restrict__ kb, __bf16* __restrict__ vtb)
{
    const int lane = threadIdx.x & 31;
    const int col  = lane & 15;
    const int h2   = lane >> 4;
    const int wid  = (blockIdx.x * blockDim.x + threadIdx.x) >> 5;
    const int mtile = wid / 48;      // 0..255 (32 rows each)
    const int nt    = wid % 48;      // 0..47  (64 cols each)

    const __bf16* arow0 = xb + (size_t)(mtile * 32 + col) * Dd;
    const __bf16* arow1 = arow0 + (size_t)16 * Dd;
    const __bf16* bptr  = wb + (size_t)(nt * 64 + col) * Dd + 16 * h2;  // f adds 16*Dd

    f32x8 acc0[4], acc1[4];
#pragma unroll
    for (int f = 0; f < 4; ++f)
#pragma unroll
        for (int r = 0; r < 8; ++r) { acc0[f][r] = 0.f; acc1[f][r] = 0.f; }

    GEMM_CORE(arow0, arow1, bptr, acc0, acc1)

#pragma unroll
    for (int f = 0; f < 4; ++f) {
        const int ng   = nt * 64 + f * 16 + col;   // 0..3071
        const float bv = bias[ng];
        const int which = ng >> 10;                // 0=q 1=k 2=v
        const int d  = ng & (Dd - 1);
        const int hh = d >> 6;
        const int dd = d & (HD - 1);
#pragma unroll
        for (int ms = 0; ms < 2; ++ms) {
#pragma unroll
            for (int r = 0; r < 8; ++r) {
                const int mg  = mtile * 32 + ms * 16 + r + 8 * h2;  // 0..8191
                const int bi  = mg >> 11;
                const int tok = mg & (Nn - 1);
                const float v = (ms ? acc1[f][r] : acc0[f][r]) + bv;
                const size_t bh = (size_t)(bi * Hh + hh);
                if (which == 0)      qb[(bh * Nn + tok) * HD + dd] = (__bf16)v;
                else if (which == 1) kb[(bh * Nn + tok) * HD + dd] = (__bf16)v;
                else                 vtb[(bh * HD + dd) * Nn + tok] = (__bf16)v;
            }
        }
    }
}

// ---------------------------------------------------------------------------
// Streaming causal attention: one wave32 per (b, h, 16-query tile).
// Online softmax over 32-key blocks; QK^T and P·V via WMMA. K/V fragments for
// the next key block are reloaded IN PLACE as soon as the current ones are
// dead (K after the score WMMAs, V after the P·V WMMAs) -> loads stay in
// flight across the softmax VALU / LDS transpose with no register copies.
// ---------------------------------------------------------------------------
__global__ __launch_bounds__(32) void attn_kernel(
    const __bf16* __restrict__ qb, const __bf16* __restrict__ kb,
    const __bf16* __restrict__ vtb, __bf16* __restrict__ attnb)
{
    __shared__ __align__(16) __bf16 Pl[16 * 32];   // P tile staging (C-frag -> A-frag)

    const int lane = threadIdx.x & 31;
    const int col  = lane & 15;
    const int h2   = lane >> 4;

    int id = blockIdx.x;
    const int qt = id & 127;  id >>= 7;
    const int h  = id & 15;   id >>= 4;
    const int b  = id;                      // 0..3
    const int q0 = qt << 4;

    const size_t bh = (size_t)(b * Hh + h);
    const __bf16* Q  = qb  + bh * (size_t)(Nn * HD);
    const __bf16* K  = kb  + bh * (size_t)(Nn * HD);
    const __bf16* VT = vtb + bh * (size_t)(HD * Nn);

    // Q A-fragments (d=0..31 and d=32..63), held for the whole loop.
    const __bf16* qrow = Q + (size_t)(q0 + col) * HD;
    const bf16x16 aq0 = ld2x8(qrow + 16 * h2,      qrow + 16 * h2 + 8);
    const bf16x16 aq1 = ld2x8(qrow + 32 + 16 * h2, qrow + 32 + 16 * h2 + 8);

    f32x8 acc[4];
    float mr[8], lr[8];
#pragma unroll
    for (int f = 0; f < 4; ++f)
#pragma unroll
        for (int r = 0; r < 8; ++r) acc[f][r] = 0.f;
#pragma unroll
    for (int r = 0; r < 8; ++r) { mr[r] = -__builtin_inff(); lr[r] = 0.f; }

    const int nblk = (q0 + 16 + 31) >> 5;   // 32-key blocks covering keys <= q0+15

    // prologue: fragments for key block 0
    bf16x16 bk[2][2], bv[4];
#pragma unroll
    for (int f = 0; f < 2; ++f) {
        const __bf16* krow = K + (size_t)(f * 16 + col) * HD + 16 * h2;
        bk[f][0] = ld2x8(krow,      krow + 8);
        bk[f][1] = ld2x8(krow + 32, krow + 40);
    }
#pragma unroll
    for (int f = 0; f < 4; ++f) {
        const __bf16* vrow = VT + (size_t)(f * 16 + col) * Nn + 16 * h2;
        bv[f] = ld2x8(vrow, vrow + 8);
    }

    for (int kbi = 0; kbi < nblk; ++kbi) {
        const int key0 = kbi << 5;
        // clamped start of the next key block (last iteration reloads current)
        const int nk0  = (kbi + 1 < nblk) ? key0 + 32 : key0;
        __builtin_prefetch(K + (size_t)(nk0 + lane) * HD, 0, 3);

        // --- scores: S[16q x 32k] as two 16x16 C-frags ---
        float s[2][8];
#pragma unroll
        for (int f = 0; f < 2; ++f) {
            f32x8 c;
#pragma unroll
            for (int r = 0; r < 8; ++r) c[r] = 0.f;
            c = wmma_bf16(aq0, bk[f][0], c);
            c = wmma_bf16(aq1, bk[f][1], c);
            const int keyg = key0 + f * 16 + col;
#pragma unroll
            for (int r = 0; r < 8; ++r) {
                const int rowg = q0 + r + 8 * h2;
                s[f][r] = (keyg <= rowg) ? c[r] * 0.125f : -__builtin_inff();
            }
        }
        // bk is dead: reload in place for the next block (in flight during
        // softmax VALU + LDS transpose + P·V WMMAs).
#pragma unroll
        for (int f = 0; f < 2; ++f) {
            const __bf16* krow = K + (size_t)(nk0 + f * 16 + col) * HD + 16 * h2;
            bk[f][0] = ld2x8(krow,      krow + 8);
            bk[f][1] = ld2x8(krow + 32, krow + 40);
        }

        // --- online softmax update (per row, across 16 lanes) ---
        float p0[8], p1[8], fac[8];
#pragma unroll
        for (int r = 0; r < 8; ++r) {
            float rm = hmax16(fmaxf(s[0][r], s[1][r]));
            float nm = fmaxf(mr[r], rm);
            fac[r] = __expf(mr[r] - nm);
            p0[r]  = __expf(s[0][r] - nm);
            p1[r]  = __expf(s[1][r] - nm);
            lr[r]  = lr[r] * fac[r] + hsum16(p0[r] + p1[r]);
            mr[r]  = nm;
        }
#pragma unroll
        for (int f = 0; f < 4; ++f)
#pragma unroll
            for (int r = 0; r < 8; ++r) acc[f][r] *= fac[r];

        // --- transpose P (C-frag layout) into A-frag layout via LDS ---
        __syncthreads();
#pragma unroll
        for (int r = 0; r < 8; ++r) {
            Pl[(r + 8 * h2) * 32 + col]      = (__bf16)p0[r];
            Pl[(r + 8 * h2) * 32 + 16 + col] = (__bf16)p1[r];
        }
        __syncthreads();
        const __bf16* prow = &Pl[col * 32];
        const bf16x16 aP = ld2x8(prow + 16 * h2, prow + 16 * h2 + 8);

        // --- acc += P(16x32) @ V(32x64): 4 WMMAs over output-column tiles ---
#pragma unroll
        for (int f = 0; f < 4; ++f) acc[f] = wmma_bf16(aP, bv[f], acc[f]);

        // bv is dead: reload in place for the next block (in flight during
        // the next block's scores + softmax).
#pragma unroll
        for (int f = 0; f < 4; ++f) {
            const __bf16* vrow = VT + (size_t)(f * 16 + col) * Nn + nk0 + 16 * h2;
            bv[f] = ld2x8(vrow, vrow + 8);
        }
    }

    // --- normalize and write merged-head bf16 activations [B][N][D] ---
#pragma unroll
    for (int f = 0; f < 4; ++f)
#pragma unroll
        for (int r = 0; r < 8; ++r) {
            const int rowg = q0 + r + 8 * h2;
            const float v = acc[f][r] / lr[r];
            attnb[((size_t)b * Nn + rowg) * Dd + h * HD + f * 16 + col] = (__bf16)v;
        }
}

// ---------------------------------------------------------------------------
// Output projection: out[m][n] = sum_k attn[m][k]*w_out[n][k] + b_out[n], fp32 out
// M=8192, N=1024, K=1024.  One wave computes a 32x64 tile.
// ---------------------------------------------------------------------------
__global__ __launch_bounds__(256) void proj_kernel(
    const __bf16* __restrict__ ab, const __bf16* __restrict__ wb,
    const float* __restrict__ bias, float* __restrict__ out)
{
    const int lane = threadIdx.x & 31;
    const int col  = lane & 15;
    const int h2   = lane >> 4;
    const int wid  = (blockIdx.x * blockDim.x + threadIdx.x) >> 5;
    const int mtile = wid >> 4;     // 0..255 (32 rows each)
    const int nt    = wid & 15;     // 0..15  (64 cols each)

    const __bf16* arow0 = ab + (size_t)(mtile * 32 + col) * Dd;
    const __bf16* arow1 = arow0 + (size_t)16 * Dd;
    const __bf16* bptr  = wb + (size_t)(nt * 64 + col) * Dd + 16 * h2;

    f32x8 acc0[4], acc1[4];
#pragma unroll
    for (int f = 0; f < 4; ++f)
#pragma unroll
        for (int r = 0; r < 8; ++r) { acc0[f][r] = 0.f; acc1[f][r] = 0.f; }

    GEMM_CORE(arow0, arow1, bptr, acc0, acc1)

#pragma unroll
    for (int f = 0; f < 4; ++f) {
        const int ng = nt * 64 + f * 16 + col;
        const float bv = bias[ng];
#pragma unroll
        for (int ms = 0; ms < 2; ++ms) {
#pragma unroll
            for (int r = 0; r < 8; ++r) {
                const int mg = mtile * 32 + ms * 16 + r + 8 * h2;
                out[(size_t)mg * Dd + ng] = (ms ? acc1[f][r] : acc0[f][r]) + bv;
            }
        }
    }
}

// ---------------------------------------------------------------------------
// Host-side launcher
// ---------------------------------------------------------------------------
extern "C" void kernel_launch(void* const* d_in, const int* in_sizes, int n_in,
                              void* d_out, int out_size, void* d_ws, size_t ws_size,
                              hipStream_t stream) {
    const float* x     = (const float*)d_in[0];   // [B,N,D]
    const float* w_qkv = (const float*)d_in[1];   // [3D,D]
    const float* b_qkv = (const float*)d_in[2];   // [3D]
    const float* w_out = (const float*)d_in[3];   // [D,D]
    const float* b_out = (const float*)d_in[4];   // [D]
    float* out = (float*)d_out;                   // [B,N,D] fp32

    char* ws = (char*)d_ws;
    // workspace layout (bytes), total ~92.3 MB (fits comfortably in 192MB L2)
    __bf16* xb    = (__bf16*)(ws + 0);           // 8192*1024  bf16 = 16 MB
    __bf16* wqkvb = (__bf16*)(ws + 16777216);    // 3072*1024  bf16 =  6 MB
    __bf16* woutb = (__bf16*)(ws + 23068672);    // 1024*1024  bf16 =  2 MB
    __bf16* qb    = (__bf16*)(ws + 25165824);    // [B*H][N][HD]    = 16 MB
    __bf16* kbuf  = (__bf16*)(ws + 41943040);    // [B*H][N][HD]    = 16 MB
    __bf16* vtb   = (__bf16*)(ws + 58720256);    // [B*H][HD][N]    = 16 MB
    __bf16* attnb = (__bf16*)(ws + 75497472);    // [B][N][D]       = 16 MB

    cvt_bf16_kernel<<<4096, 256, 0, stream>>>(x,     xb,    Bb * Nn * Dd);
    cvt_bf16_kernel<<<2048, 256, 0, stream>>>(w_qkv, wqkvb, N3 * Dd);
    cvt_bf16_kernel<<<1024, 256, 0, stream>>>(w_out, woutb, Dd * Dd);

    // 256 m-tiles * 48 n-tiles = 12288 waves / 8 waves per block
    qkv_kernel<<<1536, 256, 0, stream>>>(xb, wqkvb, b_qkv, qb, kbuf, vtb);

    // B*H*(N/16) = 8192 single-wave blocks
    attn_kernel<<<8192, 32, 0, stream>>>(qb, kbuf, vtb, attnb);

    // 256 m-tiles * 16 n-tiles = 4096 waves / 8 waves per block
    proj_kernel<<<512, 256, 0, stream>>>(attnb, woutb, b_out, out);
}
